// Attention_48309792145834
// MI455X (gfx1250) — matrix operand order repaired and checked by reading, not compiled, not run on hardware
//
#include <hip/hip_runtime.h>
#include <hip/hip_bf16.h>

// ---------------------------------------------------------------------------
// MI455X (gfx1250, wave32) attention layer:
//   qkv = x @ wqkv            -> WMMA bf16 GEMM (double-buffered LDS pipeline)
//   rope/pack                 -> q (scaled by 0.125*log2e), k, v^T as bf16
//   flash attention (causal)  -> K staged via TDM tensor_load_to_lds (D#),
//                                V^T staged via global_load_async_to_lds,
//                                S^T = K@Q^T, O^T = V^T@P^T, WMMA bf16
//   out = attn @ wo           -> WMMA bf16 GEMM (fp32 out)
// ---------------------------------------------------------------------------

#ifndef USE_ASYNC_LDS
#define USE_ASYNC_LDS 1
#endif

#if defined(__has_builtin)
#if __has_builtin(__builtin_amdgcn_tensor_load_to_lds)
#define HAVE_TDM 1
#endif
#endif
#if __has_include(<hip/amd_detail/amd_gfx1250_TDM.h>)
#define TDM_SIX_ARGS 1
#endif

typedef __attribute__((ext_vector_type(16))) __bf16 v16bf;
typedef __attribute__((ext_vector_type(8)))  float  v8f;
typedef unsigned int u32x4 __attribute__((ext_vector_type(4)));
typedef int          i32x4 __attribute__((ext_vector_type(4)));
typedef int          i32x8 __attribute__((ext_vector_type(8)));

#define S_LEN 2048
#define DIM_  2048
#define NH    32
#define NKV   8
#define HD    64
#define QKV_N 3072   // (32 + 2*8) * 64

static __device__ __forceinline__ __bf16 f2bf(float f) {
  unsigned u = __builtin_bit_cast(unsigned, f);
  unsigned short h = (unsigned short)((u + 0x7FFFu + ((u >> 16) & 1u)) >> 16);
  return __builtin_bit_cast(__bf16, h);
}
static __device__ __forceinline__ unsigned pack_bf2(float a, float b) {
  unsigned lo = (unsigned)__builtin_bit_cast(unsigned short, f2bf(a));
  unsigned hi = (unsigned)__builtin_bit_cast(unsigned short, f2bf(b));
  return lo | (hi << 16);
}

union Frag {
  v16bf v;
  uint4 u[2];
  unsigned w[8];
};

static __device__ __forceinline__ v8f wmma_bf16(v16bf a, v16bf b, v8f c) {
  return __builtin_amdgcn_wmma_f32_16x16x32_bf16(
      false, a, false, b, (short)0, c, false, false);
}

// ---- async global->LDS (ASYNCcnt path). LDS byte addr = low 32 flat bits.
static __device__ __forceinline__ void async_b128(void* lds, const void* g) {
#if USE_ASYNC_LDS
  unsigned loff = (unsigned)(size_t)lds;
  asm volatile("global_load_async_to_lds_b128 %0, %1, off"
               :: "v"(loff), "v"(g) : "memory");
#else
  *(uint4*)lds = *(const uint4*)g;
#endif
}
static __device__ __forceinline__ void async_wait0() {
#if USE_ASYNC_LDS
  asm volatile("s_wait_asynccnt 0" ::: "memory");
#endif
}

// ---- TDM (TENSORcnt path): 2D tile of 16-bit data, optional LDS row pad.
// D# encoding per CDNA5 ISA (group0: count/lds/global/type;
// group1: data_size, pad, tensor_dim0/1, tile_dim0/1, dim0 stride).
#if defined(HAVE_TDM)
static __device__ __forceinline__ void tdm_load_2d_b16(
    unsigned lds_off, const void* gaddr, unsigned tensor_d0, unsigned tensor_d1,
    unsigned tile_d0, unsigned tile_d1, unsigned stride_d0, unsigned pad_ctrl) {
  unsigned long long ga = (unsigned long long)(size_t)gaddr;
  u32x4 g0;
  g0[0] = 1u;                                  // count=1 (valid descriptor)
  g0[1] = lds_off;                             // lds_addr
  g0[2] = (unsigned)ga;                        // global_addr[31:0]
  g0[3] = ((unsigned)(ga >> 32) & 0x01FFFFFFu) // global_addr[56:32]
          | 0x80000000u;                       // type=2 ("image")
  i32x8 g1;
  g1[0] = (int)((1u << 16) | pad_ctrl);        // data_size=1 (2B) | pad bits
  g1[1] = (int)(tensor_d0 << 16);              // tensor_dim0[15:0] -> [63:48]
  g1[2] = (int)((tensor_d0 >> 16) | (tensor_d1 << 16));
  g1[3] = (int)((tensor_d1 >> 16) | (tile_d0 << 16));   // tile_dim0 [127:112]
  g1[4] = (int)(tile_d1 & 0xFFFFu);            // tile_dim1 [143:128]
  g1[5] = (int)stride_d0;                      // tensor_dim0_stride[31:0]
  g1[6] = 0;
  g1[7] = 0;
  i32x4 z4 = {0, 0, 0, 0};
#if defined(TDM_SIX_ARGS)
  i32x8 z8 = {0, 0, 0, 0, 0, 0, 0, 0};
  __builtin_amdgcn_tensor_load_to_lds(g0, g1, z4, z4, z8, 0);
#else
  __builtin_amdgcn_tensor_load_to_lds(g0, g1, z4, z4, 0);
#endif
}
static __device__ __forceinline__ void tensor_wait0() {
#if __has_builtin(__builtin_amdgcn_s_wait_tensorcnt)
  __builtin_amdgcn_s_wait_tensorcnt((short)0);
#else
  asm volatile("s_wait_tensorcnt 0" ::: "memory");
#endif
}
#endif  // HAVE_TDM

// ---------------------------------------------------------------------------
// bf16 WMMA GEMM: C[M x N] (fp32) = A[M x K] @ B[K x N]
// 256 threads (8 waves), 128x128 block tile, K-step 32, wave tile 32x64.
// Double-buffered LDS; next tile register-staged during current WMMAs.
// ---------------------------------------------------------------------------
template <int N, bool ABF16>
__global__ __launch_bounds__(256) void gemm_kernel(const void* __restrict__ Ap,
                                                   const float* __restrict__ B,
                                                   float* __restrict__ C,
                                                   int K) {
  __shared__ __bf16 lA[2][128][40];
  __shared__ __bf16 lBt[2][128][40];

  const int tid  = threadIdx.x;
  const int lane = tid & 31;
  const int wid  = tid >> 5;
  const int wm   = wid & 3;
  const int wn   = wid >> 2;
  const int m0   = blockIdx.y * 128;
  const int n0   = blockIdx.x * 128;
  const int g    = lane >> 4;
  const int ln   = lane & 15;

  v8f acc[2][4];
#pragma unroll
  for (int i = 0; i < 2; i++)
#pragma unroll
    for (int j = 0; j < 4; j++)
#pragma unroll
      for (int e = 0; e < 8; e++) acc[i][j][e] = 0.f;

  const int am  = tid >> 1;
  const int akh = (tid & 1) * 16;
  const int bn  = tid & 127;
  const int bks = (tid >> 7) * 16;
  const int nk  = K / 32;

  float arf[16];
  uint4 arh[2];
  float brf[16];

  auto gloadA = [&](int kk) {
    if constexpr (ABF16) {
      const __bf16* A = (const __bf16*)Ap;
      const uint4* src = (const uint4*)(A + (size_t)(m0 + am) * K + kk + akh);
      arh[0] = src[0];
      arh[1] = src[1];
    } else {
      const float* A = (const float*)Ap;
      const float* src = A + (size_t)(m0 + am) * K + kk + akh;
#pragma unroll
      for (int i = 0; i < 16; i++) arf[i] = src[i];
    }
  };
  auto gloadB = [&](int kk) {
#pragma unroll
    for (int i = 0; i < 16; i++)
      brf[i] = B[(size_t)(kk + bks + i) * N + n0 + bn];
  };

  gloadA(0);
  gloadB(0);

  int buf = 0;
  for (int ik = 0; ik < nk; ik++) {
    if constexpr (ABF16) {
      *(uint4*)&lA[buf][am][akh]     = arh[0];
      *(uint4*)&lA[buf][am][akh + 8] = arh[1];
    } else {
#pragma unroll
      for (int i = 0; i < 16; i++) lA[buf][am][akh + i] = f2bf(arf[i]);
    }
#pragma unroll
    for (int i = 0; i < 16; i++) lBt[buf][bn][bks + i] = f2bf(brf[i]);
    __syncthreads();

    if (ik + 1 < nk) {
      gloadA((ik + 1) * 32);
      gloadB((ik + 1) * 32);
      if (ik + 2 < nk) {
        __builtin_prefetch(&B[(size_t)((ik + 2) * 32 + bks) * N + n0 + bn], 0, 1);
        if constexpr (!ABF16)
          __builtin_prefetch((const float*)Ap + (size_t)(m0 + am) * K +
                                 (ik + 2) * 32 + akh, 0, 1);
      }
    }

    Frag a[2], b4[4];
#pragma unroll
    for (int mt = 0; mt < 2; mt++) {
      const __bf16* base = &lA[buf][wm * 32 + mt * 16 + ln][0];
      a[mt].u[0] = *(const uint4*)(base + 8 * g);
      a[mt].u[1] = *(const uint4*)(base + 16 + 8 * g);
    }
#pragma unroll
    for (int nt = 0; nt < 4; nt++) {
      const __bf16* base = &lBt[buf][wn * 64 + nt * 16 + ln][0];
      b4[nt].u[0] = *(const uint4*)(base + 8 * g);
      b4[nt].u[1] = *(const uint4*)(base + 16 + 8 * g);
    }
#pragma unroll
    for (int mt = 0; mt < 2; mt++)
#pragma unroll
      for (int nt = 0; nt < 4; nt++)
        acc[mt][nt] = wmma_bf16(a[mt].v, b4[nt].v, acc[mt][nt]);

    buf ^= 1;
  }

#pragma unroll
  for (int mt = 0; mt < 2; mt++)
#pragma unroll
    for (int nt = 0; nt < 4; nt++) {
      const int col = n0 + wn * 64 + nt * 16 + ln;
#pragma unroll
      for (int r = 0; r < 8; r++) {
        const int row = m0 + wm * 32 + mt * 16 + r + 8 * g;
        C[(size_t)row * N + col] = acc[mt][nt][r];
      }
    }
}

// ---------------------------------------------------------------------------
// RoPE + repack + bf16 convert.
// ---------------------------------------------------------------------------
__global__ __launch_bounds__(128) void rope_pack_kernel(
    const float* __restrict__ qkv, const float* __restrict__ cosp,
    const float* __restrict__ sinp, __bf16* __restrict__ qb,
    __bf16* __restrict__ kb, __bf16* __restrict__ vt) {
  const int s = blockIdx.x;
  const float* row = qkv + (size_t)s * QKV_N;
  const float QSCALE = 0.18033688011112042f;  // (1/sqrt(64)) * log2(e)

  for (int c = threadIdx.x; c < QKV_N; c += 128) {
    if (c < DIM_) {
      const int h = c >> 6, d = c & 63;
      float v = row[c];
      if (d < 32) {
        const float cs = cosp[s * 32 + d], sn = sinp[s * 32 + d];
        const float other = (d < 16) ? -row[c + 16] : row[c - 16];
        v = v * cs + other * sn;
      }
      qb[((size_t)h * S_LEN + s) * HD + d] = f2bf(v * QSCALE);
    } else if (c < DIM_ + NKV * HD) {
      const int cc = c - DIM_;
      const int kh = cc >> 6, d = cc & 63;
      float v = row[c];
      if (d < 32) {
        const float cs = cosp[s * 32 + d], sn = sinp[s * 32 + d];
        const float other = (d < 16) ? -row[c + 16] : row[c - 16];
        v = v * cs + other * sn;
      }
      kb[((size_t)kh * S_LEN + s) * HD + d] = f2bf(v);
    } else {
      const int cc = c - DIM_ - NKV * HD;
      const int kh = cc >> 6, d = cc & 63;
      vt[((size_t)kh * HD + d) * S_LEN + s] = f2bf(row[c]);
    }
  }
}

// ---------------------------------------------------------------------------
// Causal flash attention.
// Workgroup = 4 waves = the 4 Q heads of one KV head, one 16-row q block.
// Per 32-key block: K tile (32x64) arrives via a single TDM descriptor load
// (wave 0, TENSORcnt), V^T tile (64x32) via per-lane async-to-LDS (ASYNCcnt);
// both double-buffered and issued one block ahead of the WMMAs.
// ---------------------------------------------------------------------------
__global__ __launch_bounds__(128) void attn_kernel(const __bf16* __restrict__ qb,
                                                   const __bf16* __restrict__ kb,
                                                   const __bf16* __restrict__ vt,
                                                   __bf16* __restrict__ ao) {
  __shared__ __bf16 lK[2][32][72];   // [key][dim], 144B rows (128B + 16B pad)
  __shared__ __bf16 lV[2][64][40];   // [dim][key], 80B rows

  const int qblk = blockIdx.x;
  const int kv   = blockIdx.y;
  const int tid  = threadIdx.x;
  const int wid  = tid >> 5;
  const int lane = tid & 31;
  const int h    = kv * 4 + wid;
  const int g    = lane >> 4;
  const int ln   = lane & 15;
  const int qrow = qblk * 16 + ln;

  auto stage = [&](int kbase, int bsel) {
#if defined(HAVE_TDM)
    if (wid == 0) {
      // pad: 32 DWORDs (128B row) then 4 DWORDs (16B) -> codes 4 and 3
      const unsigned pad_ctrl = (1u << 20) | (4u << 22) | (3u << 25);
      tdm_load_2d_b16((unsigned)(size_t)&lK[bsel][0][0],
                      kb + ((size_t)kv * S_LEN + kbase) * HD,
                      /*tensor_d0=*/HD, /*tensor_d1=*/S_LEN,
                      /*tile_d0=*/HD, /*tile_d1=*/32,
                      /*stride_d0=*/HD, pad_ctrl);
    }
#else
#pragma unroll
    for (int j = 0; j < 2; j++) {
      const int ck = tid * 2 + j;
      const int krow = ck >> 3, kch = ck & 7;
      async_b128(&lK[bsel][krow][kch * 8],
                 kb + ((size_t)kv * S_LEN + kbase + krow) * HD + kch * 8);
    }
#endif
#pragma unroll
    for (int j = 0; j < 2; j++) {
      const int ck = tid * 2 + j;
      const int vrow = ck >> 2, vch = ck & 3;
      async_b128(&lV[bsel][vrow][vch * 8],
                 vt + ((size_t)kv * HD + vrow) * S_LEN + kbase + vch * 8);
    }
  };

  // Q^T B-fragments (loaded once from global)
  Frag bq[2];
#pragma unroll
  for (int grp = 0; grp < 2; grp++) {
    const __bf16* p = qb + ((size_t)h * S_LEN + qrow) * HD + grp * 32;
    bq[grp].u[0] = *(const uint4*)(p + 8 * g);
    bq[grp].u[1] = *(const uint4*)(p + 16 + 8 * g);
  }

  v8f o[4];
#pragma unroll
  for (int i = 0; i < 4; i++)
#pragma unroll
    for (int e = 0; e < 8; e++) o[i][e] = 0.f;
  float m = -1e30f, l = 0.f;

  const int nkb = (qblk >> 1) + 1;
  stage(0, 0);
  int buf = 0;

  for (int ib = 0; ib < nkb; ib++) {
    const int kbase = ib * 32;
    async_wait0();
#if defined(HAVE_TDM)
    if (wid == 0) tensor_wait0();
#endif
    __syncthreads();
    if (ib + 1 < nkb) stage((ib + 1) * 32, buf ^ 1);

    Frag ak[2][2];
#pragma unroll
    for (int t = 0; t < 2; t++) {
      const __bf16* krow = &lK[buf][t * 16 + ln][0];
#pragma unroll
      for (int grp = 0; grp < 2; grp++) {
        ak[t][grp].u[0] = *(const uint4*)(krow + grp * 32 + 8 * g);
        ak[t][grp].u[1] = *(const uint4*)(krow + grp * 32 + 16 + 8 * g);
      }
    }

    v8f st[2];
#pragma unroll
    for (int t = 0; t < 2; t++) {
      v8f z;
#pragma unroll
      for (int e = 0; e < 8; e++) z[e] = 0.f;
      z     = wmma_bf16(ak[t][0].v, bq[0].v, z);
      st[t] = wmma_bf16(ak[t][1].v, bq[1].v, z);
    }

    if (ib == nkb - 1) {
#pragma unroll
      for (int t = 0; t < 2; t++)
#pragma unroll
        for (int r = 0; r < 8; r++) {
          const int key = kbase + t * 16 + r + 8 * g;
          if (key > qrow) st[t][r] = -1e30f;
        }
    }

    // online softmax (log2 domain; raw v_exp_f32)
    float vmax = -1e30f;
#pragma unroll
    for (int t = 0; t < 2; t++)
#pragma unroll
      for (int r = 0; r < 8; r++) vmax = fmaxf(vmax, st[t][r]);
    vmax = fmaxf(vmax, __shfl_xor(vmax, 16, 32));
    const float mn    = fmaxf(m, vmax);
    const float alpha = __builtin_amdgcn_exp2f(m - mn);
    float p[16];
    float psum = 0.f;
#pragma unroll
    for (int t = 0; t < 2; t++)
#pragma unroll
      for (int r = 0; r < 8; r++) {
        const float e = __builtin_amdgcn_exp2f(st[t][r] - mn);
        p[t * 8 + r] = e;
        psum += e;
      }
    psum += __shfl_xor(psum, 16, 32);
    l = l * alpha + psum;
    m = mn;
#pragma unroll
    for (int i = 0; i < 4; i++)
#pragma unroll
      for (int e = 0; e < 8; e++) o[i][e] *= alpha;

    // P^T B-fragment: pack to bf16 pairs first, then exchange packed dwords
    // (8 ds_bpermute instead of 16)
    unsigned pk0[4], pk1[4];
#pragma unroll
    for (int i = 0; i < 4; i++) {
      pk0[i] = pack_bf2(p[2 * i], p[2 * i + 1]);
      pk1[i] = pack_bf2(p[8 + 2 * i], p[9 + 2 * i]);
    }
    Frag bp;
#pragma unroll
    for (int i = 0; i < 4; i++) {
      const unsigned op0 = __shfl_xor(pk0[i], 16, 32);
      const unsigned op1 = __shfl_xor(pk1[i], 16, 32);
      bp.w[i]     = (g == 0) ? pk0[i] : op1;  // keys 0-7  / 16-23
      bp.w[4 + i] = (g == 0) ? op0 : pk1[i];  // keys 8-15 / 24-31
    }

    // O^T += V^T @ P^T over 4 dim tiles (V^T from LDS)
#pragma unroll
    for (int dt = 0; dt < 4; dt++) {
      Frag av;
      const __bf16* vrow = &lV[buf][dt * 16 + ln][0];
      av.u[0] = *(const uint4*)(vrow + 16 * g);
      av.u[1] = *(const uint4*)(vrow + 16 * g + 8);
      o[dt] = wmma_bf16(av.v, bp.v, o[dt]);
    }
    buf ^= 1;
  }

  const float inv = __builtin_amdgcn_rcpf(l);
#pragma unroll
  for (int dt = 0; dt < 4; dt++) {
#pragma unroll
    for (int r = 0; r < 8; r += 2) {
      const int col = h * 64 + dt * 16 + 8 * g + r;
      *(unsigned*)(ao + (size_t)qrow * DIM_ + col) =
          pack_bf2(o[dt][r] * inv, o[dt][r + 1] * inv);
    }
  }
}

// ---------------------------------------------------------------------------
extern "C" void kernel_launch(void* const* d_in, const int* in_sizes, int n_in,
                              void* d_out, int out_size, void* d_ws,
                              size_t ws_size, hipStream_t stream) {
  (void)in_sizes; (void)n_in; (void)out_size; (void)ws_size;

  const float* x    = (const float*)d_in[0];
  const float* cosp = (const float*)d_in[1];
  const float* sinp = (const float*)d_in[2];
  const float* wqkv = (const float*)d_in[3];
  const float* wo   = (const float*)d_in[4];
  float* out = (float*)d_out;

  char* ws = (char*)d_ws;
  float*  qkv = (float*)ws;                                               // 24 MB
  __bf16* qb  = (__bf16*)(ws + (size_t)25165824);                         //  8 MB
  __bf16* kb  = (__bf16*)(ws + (size_t)25165824 + 8388608);               //  2 MB
  __bf16* vt  = (__bf16*)(ws + (size_t)25165824 + 8388608 + 2097152);     //  2 MB
  __bf16* ao  = (__bf16*)(ws + (size_t)25165824 + 8388608 + 2 * 2097152); //  8 MB

  gemm_kernel<QKV_N, false>
      <<<dim3(QKV_N / 128, S_LEN / 128), 256, 0, stream>>>(x, wqkv, qkv, DIM_);

  rope_pack_kernel<<<S_LEN, 128, 0, stream>>>(qkv, cosp, sinp, qb, kb, vt);

  attn_kernel<<<dim3(S_LEN / 16, NKV), 128, 0, stream>>>(qb, kb, vt, ao);

  gemm_kernel<DIM_, true>
      <<<dim3(DIM_ / 128, S_LEN / 128), 256, 0, stream>>>(ao, wo, out, NH * HD);
}
